// NonLocalBlock_12463995093701
// MI455X (gfx1250) — matrix operand order repaired and checked by reading, not compiled, not run on hardware
//
#include <hip/hip_runtime.h>
#include <hip/hip_bf16.h>

#define C_CH 256
#define NSP  4096
#define NB   4
#define GRP  32
#define CPG  8
#define JBK  128

typedef __attribute__((ext_vector_type(16))) _Float16 v16h;
typedef __attribute__((ext_vector_type(8)))  _Float16 v8h;
typedef __attribute__((ext_vector_type(8)))  float    v8f;
typedef __attribute__((ext_vector_type(4)))  unsigned int v4u;
typedef __attribute__((ext_vector_type(8)))  int v8i;
typedef __attribute__((ext_vector_type(4)))  int v4i;

union F8 { v8f v; float f[8]; };

__device__ __forceinline__ v16h cat16(v8h lo, v8h hi) {
  union { v16h v; v8h h[2]; } u; u.h[0] = lo; u.h[1] = hi; return u.v;
}
// A operand (16x32 f16): lane holds row M=lane%16; lane-half selects K {0..7,16..23} / {8..15,24..31}
__device__ __forceinline__ v16h load_a(const _Float16* rowp, int k0, int lh) {
  const _Float16* p = rowp + k0 + lh * 8;
  return cat16(*(const v8h*)p, *(const v8h*)(p + 16));
}
// B operand (32x16 f16): lane holds column N=lane%16; lane-half selects K {0..7,16..23} / {8..15,24..31}
__device__ __forceinline__ v16h load_b(const _Float16* colp, int k0, int lh) {
  const _Float16* p = colp + k0 + lh * 8;
  return cat16(*(const v8h*)p, *(const v8h*)(p + 16));
}
__device__ __forceinline__ v8f wmma16(v16h a, v16h b, v8f c) {
  return __builtin_amdgcn_wmma_f32_16x16x32_f16(false, a, false, b, (short)0, c, false, false);
}
// LDS byte offset of a __shared__ object (generic -> AS(3) -> int)
__device__ __forceinline__ unsigned int lds_offset_of(const void* p) {
  return (unsigned int)(unsigned long long)(__attribute__((address_space(3))) const void*)p;
}

// ---------------- GroupNorm -> hnT (N-major f16) ----------------
__global__ void gn_hn_kernel(const float* __restrict__ x, const float* __restrict__ gam,
                             const float* __restrict__ bet, _Float16* __restrict__ hnT) {
  int b = blockIdx.x >> 5, g = blockIdx.x & 31, tid = threadIdx.x;
  const float* xg = x + ((size_t)(b * C_CH + g * CPG) << 12);
  float s = 0.f, ss = 0.f;
  for (int i = tid; i < CPG * NSP; i += 256) { float v = xg[i]; s += v; ss += v * v; }
  __shared__ float rs[256], rss[256];
  rs[tid] = s; rss[tid] = ss; __syncthreads();
  for (int o = 128; o > 0; o >>= 1) {
    if (tid < o) { rs[tid] += rs[tid + o]; rss[tid] += rss[tid + o]; }
    __syncthreads();
  }
  float mean = rs[0] * (1.f / 32768.f);
  float var  = rss[0] * (1.f / 32768.f) - mean * mean;
  float rinv = rsqrtf(var + 1e-6f);
  for (int i = tid; i < CPG * NSP; i += 256) {
    int cl = i >> 12, n = i & 4095, c = g * CPG + cl;
    float xn = (xg[i] - mean) * rinv * gam[c] + bet[c];
    hnT[(((size_t)b * NSP + n) << 8) + c] = (_Float16)xn;
  }
}

// ---------------- weights fp32 -> f16 (wq|wk|wv|wp packed) ----------------
__global__ void wcvt_kernel(const float* __restrict__ wq, const float* __restrict__ wk,
                            const float* __restrict__ wv, const float* __restrict__ wp,
                            _Float16* __restrict__ w16) {
  int i = blockIdx.x * 256 + threadIdx.x;
  int sel = i >> 16, j = i & 65535;
  const float* s = (sel == 0) ? wq : (sel == 1) ? wk : (sel == 2) ? wv : wp;
  w16[i] = (_Float16)s[j];
}

// ---------------- QKV projection GEMMs (WMMA), shared B-operand ----------------
__global__ void qkv_kernel(const _Float16* __restrict__ hnT, const _Float16* __restrict__ w16,
                           const float* __restrict__ bq, const float* __restrict__ bk,
                           const float* __restrict__ bv,
                           _Float16* __restrict__ qT, _Float16* __restrict__ kT,
                           _Float16* __restrict__ vv) {
  int wid = blockIdx.x * 8 + (threadIdx.x >> 5);
  int lane = threadIdx.x & 31, lh = lane >> 4, li = lane & 15;
  int d0 = (wid & 15) << 4;
  int n0 = (wid >> 4) << 4;
  const _Float16* wqa = w16 + (size_t)(d0 + li) * C_CH;
  const _Float16* wka = wqa + 65536;
  const _Float16* wva = wqa + 131072;
  const _Float16* hb  = hnT + ((size_t)(n0 + li) << 8);
  F8 aq{}, ak{}, av{};
  #pragma unroll
  for (int kc = 0; kc < C_CH; kc += 32) {
    v16h bop = load_b(hb, kc, lh);
    aq.v = wmma16(load_a(wqa, kc, lh), bop, aq.v);
    ak.v = wmma16(load_a(wka, kc, lh), bop, ak.v);
    av.v = wmma16(load_a(wva, kc, lh), bop, av.v);
  }
  int n = n0 + li, bb = n >> 12, nn = n & 4095;
  #pragma unroll
  for (int m = 0; m < 8; m++) {
    int d = d0 + m + 8 * lh;
    qT[((size_t)n << 8) + d] = (_Float16)(aq.f[m] + bq[d]);
    kT[((size_t)n << 8) + d] = (_Float16)(ak.f[m] + bk[d]);
    vv[(((size_t)bb * C_CH + d) << 12) + nn] = (_Float16)(av.f[m] + bv[d]);
  }
}

// ---------------- flash attention: softmax(q.k/16) applied to v ----------------
__global__ void attn_kernel(const _Float16* __restrict__ qT, const _Float16* __restrict__ kT,
                            const _Float16* __restrict__ vv, _Float16* __restrict__ AT) {
  __shared__ _Float16 qlds[16 * 264];   // padded stride (264 halfs): 16B aligned, bank-spread
  __shared__ _Float16 plds[16 * 136];   // P^T in B-operand layout
  __shared__ float wmx[128], wsm[128], m_run[16], l_run[16], alp[16], mnw[16], lnv[16];
  int tid = threadIdx.x, wvi = tid >> 5, lane = tid & 31, lh = lane >> 4, li = lane & 15;
  int b = blockIdx.x >> 8, it = (blockIdx.x & 255) << 4;
  const _Float16* qb = qT + (((size_t)b * NSP + it) << 8);

  // --- TDM: DMA the 16x256 f16 q tile into LDS with row padding (512B data + 16B pad) ---
  if (wvi == 0) {
    unsigned long long ga = (unsigned long long)(const void*)qb;
    v4u g0;
    g0[0] = 1u;                               // count=1, user descriptor
    g0[1] = lds_offset_of(qlds);              // lds_addr (bytes)
    g0[2] = (unsigned int)ga;                 // global_addr[31:0]
    g0[3] = (unsigned int)((ga >> 32) & 0x1FFFFFFu) | (2u << 30);  // addr[56:32] | type=2
    v8i g1;
    g1[0] = (1 << 16)        // data_size = 2 bytes
          | (1 << 20)        // pad_enable
          | (6 << 22)        // pad_interval: 128 DWORDs (512B) between pads
          | (3 << 25);       // pad_amount: 4 DWORDs (16B)
    g1[1] = (int)(256u << 16);  // tensor_dim0[15:0] = 256 (bits 79:48, low half here)
    g1[2] = (int)(16u << 16);   // tensor_dim1 = 16  (bits 111:80, low half here)
    g1[3] = (int)(256u << 16);  // tile_dim0 = 256   (bits 127:112)
    g1[4] = 16;                 // tile_dim1 = 16    (bits 143:128)
    g1[5] = 256;                // tensor_dim0_stride = 256 (bits 207:160, low 32)
    g1[6] = 0;
    g1[7] = 0;
    v4i gz4 = {0, 0, 0, 0};
    v8i gz8 = {0, 0, 0, 0, 0, 0, 0, 0};
    __builtin_amdgcn_tensor_load_to_lds(g0, g1, gz4, gz4, gz8, 0);
    __builtin_amdgcn_s_wait_tensorcnt(0);
  }
  if (tid < 16) { m_run[tid] = -3.0e30f; l_run[tid] = 0.f; }
  __syncthreads();

  // hoist q B-operand into registers (read LDS once)
  v16h qreg[8];
  #pragma unroll
  for (int k8 = 0; k8 < 8; k8++) qreg[k8] = load_b(qlds + li * 264, k8 * 32, lh);

  F8 O0{}, O1{};
  const _Float16* kb  = kT + ((size_t)b * NSP << 8);
  const _Float16* vb  = vv + ((size_t)b * C_CH << 12);
  const _Float16* va0 = vb + ((size_t)(wvi * 32 + li) << 12);
  const _Float16* va1 = va0 + ((size_t)16 << 12);
  for (int jb = 0; jb < NSP; jb += JBK) {
    const _Float16* ka = kb + ((size_t)(jb + wvi * 16 + li) << 8);
    if (jb + JBK < NSP) {   // prefetch next block's K/V rows while this block computes
      __builtin_prefetch(ka + (JBK << 8), 0, 1);
      __builtin_prefetch(va0 + jb + JBK, 0, 1);
      __builtin_prefetch(va1 + jb + JBK, 0, 1);
    }
    // scores S^T (16 j-rows x 16 i-cols), K = C = 256
    F8 s{};
    #pragma unroll
    for (int k8 = 0; k8 < 8; k8++)
      s.v = wmma16(load_a(ka, k8 * 32, lh), qreg[k8], s.v);
    float mx = -3.0e30f;
    #pragma unroll
    for (int m = 0; m < 8; m++) { s.f[m] *= 0.0625f; mx = fmaxf(mx, s.f[m]); }
    mx = fmaxf(mx, __shfl_xor(mx, 16, 32));
    if (lane < 16) wmx[wvi * 16 + lane] = mx;
    __syncthreads();
    if (tid < 16) {
      float bm = wmx[tid];
      #pragma unroll
      for (int w = 1; w < 8; w++) bm = fmaxf(bm, wmx[w * 16 + tid]);
      float mo = m_run[tid], mn2 = fmaxf(mo, bm);
      mnw[tid] = mn2; alp[tid] = __expf(mo - mn2); m_run[tid] = mn2;
    }
    __syncthreads();
    float mn = mnw[li], al = alp[li], sm = 0.f;
    #pragma unroll
    for (int m = 0; m < 8; m++) {
      float p = __expf(s.f[m] - mn); sm += p;
      plds[li * 136 + wvi * 16 + m + 8 * lh] = (_Float16)p;
    }
    sm += __shfl_xor(sm, 16, 32);
    if (lane < 16) wsm[wvi * 16 + lane] = sm;
    #pragma unroll
    for (int m = 0; m < 8; m++) { O0.f[m] *= al; O1.f[m] *= al; }
    __syncthreads();
    if (tid < 16) {
      float t = 0.f;
      #pragma unroll
      for (int w = 0; w < 8; w++) t += wsm[w * 16 + tid];
      l_run[tid] = l_run[tid] * alp[tid] + t;
    }
    // apply: O(c,i) += V(c,j) * P^T(j,i), wave owns 32 channels
    #pragma unroll
    for (int jt = 0; jt < JBK; jt += 32) {
      v16h bP = load_b(plds + li * 136, jt, lh);
      O0.v = wmma16(load_a(va0, jb + jt, lh), bP, O0.v);
      O1.v = wmma16(load_a(va1, jb + jt, lh), bP, O1.v);
    }
    __syncthreads();
  }
  if (tid < 16) lnv[tid] = 1.0f / l_run[tid];
  __syncthreads();
  float inv = lnv[li];
  _Float16* ab = AT + (((size_t)b * NSP + it + li) << 8) + wvi * 32;
  #pragma unroll
  for (int m = 0; m < 8; m++) {
    ab[m + 8 * lh]      = (_Float16)(O0.f[m] * inv);
    ab[16 + m + 8 * lh] = (_Float16)(O1.f[m] * inv);
  }
}

// ---------------- output projection + residual (fp32 out) ----------------
__global__ void proj_kernel(const _Float16* __restrict__ AT, const _Float16* __restrict__ wp16,
                            const float* __restrict__ bp, const float* __restrict__ x,
                            float* __restrict__ out) {
  int wid = blockIdx.x * 8 + (threadIdx.x >> 5);
  int lane = threadIdx.x & 31, lh = lane >> 4, li = lane & 15;
  int d0 = (wid & 15) << 4, n0 = (wid >> 4) << 4;
  const _Float16* wa = wp16 + (size_t)(d0 + li) * C_CH;
  const _Float16* ab = AT + ((size_t)(n0 + li) << 8);
  F8 acc{};
  #pragma unroll
  for (int kc = 0; kc < C_CH; kc += 32)
    acc.v = wmma16(load_a(wa, kc, lh), load_b(ab, kc, lh), acc.v);
  int n = n0 + li, bb = n >> 12, nn = n & 4095;
  #pragma unroll
  for (int m = 0; m < 8; m++) {
    int d = d0 + m + 8 * lh;
    size_t oi = (((size_t)bb * C_CH + d) << 12) + nn;
    out[oi] = x[oi] + acc.f[m] + bp[d];
  }
}

extern "C" void kernel_launch(void* const* d_in, const int* in_sizes, int n_in,
                              void* d_out, int out_size, void* d_ws, size_t ws_size,
                              hipStream_t stream) {
  (void)in_sizes; (void)n_in; (void)out_size; (void)ws_size;
  const float* x   = (const float*)d_in[0];
  const float* gns = (const float*)d_in[1];
  const float* gnb = (const float*)d_in[2];
  const float* wq  = (const float*)d_in[3];
  const float* bq  = (const float*)d_in[4];
  const float* wk  = (const float*)d_in[5];
  const float* bk  = (const float*)d_in[6];
  const float* wv  = (const float*)d_in[7];
  const float* bv  = (const float*)d_in[8];
  const float* wp  = (const float*)d_in[9];
  const float* bp  = (const float*)d_in[10];
  float* out = (float*)d_out;
  char* ws = (char*)d_ws;
  const size_t SZ = (size_t)NB * NSP * C_CH * sizeof(_Float16); // 8 MiB per f16 tensor
  _Float16* hnT = (_Float16*)(ws);
  _Float16* qT  = (_Float16*)(ws + SZ);
  _Float16* kT  = (_Float16*)(ws + 2 * SZ);
  _Float16* vv  = (_Float16*)(ws + 3 * SZ);
  _Float16* AT  = (_Float16*)(ws + 4 * SZ);
  _Float16* w16 = (_Float16*)(ws + 5 * SZ);

  gn_hn_kernel<<<NB * GRP, 256, 0, stream>>>(x, gns, gnb, hnT);
  wcvt_kernel<<<1024, 256, 0, stream>>>(wq, wk, wv, wp, w16);
  qkv_kernel<<<2048, 256, 0, stream>>>(hnT, w16, bq, bk, bv, qT, kT, vv);
  attn_kernel<<<NB * (NSP / 16), 256, 0, stream>>>(qT, kT, vv, AT);
  proj_kernel<<<2048, 256, 0, stream>>>(AT, w16 + 3 * 65536, bp, x, out);
}
